// MemoryBankContrastLoss_19636590477428
// MI455X (gfx1250) — matrix operand order repaired and checked by reading, not compiled, not run on hardware
//
#include <hip/hip_runtime.h>
#include <hip/hip_bf16.h>

// ---------------------------------------------------------------------------
// MemoryBankContrastLoss for MI455X (gfx1250, wave32)
//   main_proj [4,256,128,128] f32, main_gt [4,512,512] int,
//   aux_proj  [4,256,128,128] f32, aux_gt  [4,512,512] int,
//   memory_bank [20,512,256] f32  ->  scalar f32 loss
// Heavy op: 10240x10240x256 GEMM fused with masked log-softmax, done with
// v_wmma_f32_16x16x32_bf16 using a hi/lo bf16 split of fp32 operands.
// Mapping: one 16-row tile per block; the 4 waves partition the 640 column
// tiles (no redundant B traffic, 2560 waves total); fixed-shift online
// softmax partials merged through LDS. B stream prefetched 2 tiles ahead
// with unconditional speculative global_prefetch_b8.
// ---------------------------------------------------------------------------

typedef __bf16 v16bf __attribute__((ext_vector_type(16)));
typedef __bf16 v8bf  __attribute__((ext_vector_type(8)));
typedef float  v8f   __attribute__((ext_vector_type(8)));

#define NCLS   20
#define MEMM   512
#define DDIM   256
#define NPIX   131072          // 4 * 2 * 128*128
#define AROWS  (NCLS * MEMM)   // 10240
#define NCT    (AROWS / 16)    // 640 column tiles
#define EPSV   1e-12f
#define INVT   10.0f           // 1 / TEMP

// ---------------------------------------------------------------------------
// K1: per class, find the first MEMM pixel indices (ascending) whose label==c.
// One block per class, 1024 threads, ballot-based stable block prefix scan.
// ---------------------------------------------------------------------------
__global__ __launch_bounds__(1024) void mbcl_select(const int* __restrict__ mgt,
                                                    const int* __restrict__ agt,
                                                    int* __restrict__ order,
                                                    int* __restrict__ counts) {
  __shared__ unsigned wsum[32];
  __shared__ unsigned woff[33];
  const int c    = blockIdx.x;
  const int tid  = threadIdx.x;
  const int lane = tid & 31;
  const int wv   = tid >> 5;
  unsigned base = 0;
  for (int chunk = 0; chunk < NPIX / 1024; ++chunk) {
    const int n  = chunk * 1024 + tid;
    const int b  = n >> 15;           // batch
    const int r  = n & 32767;
    const int br = r >> 14;           // 0 = main, 1 = aux
    const int p  = r & 16383;         // i*128 + j in 128x128
    const int i  = p >> 7;
    const int j  = p & 127;
    const int* gt = br ? agt : mgt;
    int lab = gt[(size_t)b * 262144 + (size_t)(i * 4) * 512 + (size_t)(j * 4)];
    const bool valid = (lab != 255);
    if (lab == 254) lab = NCLS - 1;
    const bool match = valid && (lab == c);

    const unsigned mask = (unsigned)__ballot(match);
    const unsigned pre  = __popc(mask & ((1u << lane) - 1u));
    if (lane == 0) wsum[wv] = __popc(mask);
    __syncthreads();
    if (tid == 0) {
      unsigned s = 0;
      for (int w = 0; w < 32; ++w) { woff[w] = s; s += wsum[w]; }
      woff[32] = s;
    }
    __syncthreads();
    const unsigned rank = base + woff[wv] + pre;
    if (match && rank < (unsigned)MEMM) order[c * MEMM + rank] = n;
    base += woff[32];
    __syncthreads();
    if (base >= (unsigned)MEMM) break;   // uniform: enough samples found
  }
  if (tid == 0) counts[c] = (int)base;   // >= MEMM if early-exited (sufficient)
}

// ---------------------------------------------------------------------------
// K2: one block (256 thr) per (class,slot) row.
//   anchors = L2-normalized gathered feature; bank' = renorm(0.999*bank+0.001*sel)
//   Emit bf16 hi/lo splits of both + a_valid / c_valid flags.
// ---------------------------------------------------------------------------
__device__ __forceinline__ float block_sum256(float v, float* red) {
  const int t = threadIdx.x;
  red[t] = v; __syncthreads();
  for (int s = 128; s > 0; s >>= 1) {
    if (t < s) red[t] += red[t + s];
    __syncthreads();
  }
  const float r = red[0];
  __syncthreads();
  return r;
}

__global__ __launch_bounds__(256) void mbcl_gather(const float* __restrict__ mainp,
                                                   const float* __restrict__ auxp,
                                                   const float* __restrict__ bank,
                                                   const int* __restrict__ order,
                                                   const int* __restrict__ counts,
                                                   float* __restrict__ a_valid,
                                                   float* __restrict__ c_valid,
                                                   __bf16* __restrict__ a_hi,
                                                   __bf16* __restrict__ a_lo,
                                                   __bf16* __restrict__ b_hi,
                                                   __bf16* __restrict__ b_lo) {
  __shared__ float red[256];
  const int row = blockIdx.x;
  const int cls = row >> 9;
  const int m   = row & (MEMM - 1);
  const int d   = threadIdx.x;
  const bool valid = m < counts[cls];          // uniform per block

  float x = 0.0f;
  if (valid) {
    const int n  = order[row];
    const int b  = n >> 15;
    const int r  = n & 32767;
    const int br = r >> 14;
    const int p  = r & 16383;
    const float* src = br ? auxp : mainp;
    x = src[(size_t)b * 4194304 + (size_t)d * 16384 + (size_t)p];
  }
  const float n2s = block_sum256(x * x, red);
  const float sel = x * (1.0f / fmaxf(sqrtf(n2s), EPSV));

  // anchors (zero rows when invalid)
  const float ahf = (__bf16)sel;
  a_hi[(size_t)row * DDIM + d] = (__bf16)sel;
  a_lo[(size_t)row * DDIM + d] = (__bf16)(sel - ahf);

  // bank momentum update
  const float bk = bank[(size_t)row * DDIM + d];
  float fin = bk;
  if (valid) {
    float u = 0.999f * bk + 0.001f * sel;
    const float n2u = block_sum256(u * u, red);
    fin = u * (1.0f / fmaxf(sqrtf(n2u), EPSV));
  }
  const float bhf = (__bf16)fin;
  b_hi[(size_t)row * DDIM + d] = (__bf16)fin;
  b_lo[(size_t)row * DDIM + d] = (__bf16)(fin - bhf);

  const float n2f = block_sum256(fin * fin, red);  // c_valid = ||bank row|| > 0
  if (d == 0) {
    a_valid[row] = valid ? 1.0f : 0.0f;
    c_valid[row] = (n2f > 0.0f) ? 1.0f : 0.0f;
  }
}

// ---------------------------------------------------------------------------
// K3: fused GEMM + masked online log-softmax loss pieces.
// One 16-row tile per block; 4 waves partition the 640 column tiles (160 each).
// 24 v_wmma_f32_16x16x32_bf16 per tile (hi*hi + hi*lo + lo*hi).
// Fixed shift (scores <= 1/TEMP = 10 for unit vectors) makes cross-wave merge
// a pure add of (sum-exp, pos-sum, count) partials through LDS.
// ---------------------------------------------------------------------------
__global__ __launch_bounds__(128) void mbcl_gemm_loss(const __bf16* __restrict__ a_hi,
                                                      const __bf16* __restrict__ a_lo,
                                                      const __bf16* __restrict__ b_hi,
                                                      const __bf16* __restrict__ b_lo,
                                                      const float* __restrict__ a_valid,
                                                      const float* __restrict__ c_valid,
                                                      float* __restrict__ mlpp,
                                                      float* __restrict__ varr) {
  __shared__ float s_l[4][16];
  __shared__ float s_p[4][16];
  __shared__ float s_c[4];

  const int wave = threadIdx.x >> 5;
  const int lane = threadIdx.x & 31;
  const int half = lane >> 4;          // 0: K-low halves, 1: K-high halves
  const int l16  = lane & 15;
  const int rt   = blockIdx.x * 16;    // this block's row tile
  const int myclass = rt >> 9;

  // --- hoist the A fragments (16 rows x 256 K, hi+lo) into registers ---
  // A layout (matches B): lane half selects K base (0/16), elems 0-15 at
  // K=base..base+15 within each 32-K chunk -> two 16B contiguous loads.
  const int arow  = rt + l16;
  const int abase = half * 16;
  const __bf16* aph = a_hi + (size_t)arow * DDIM;
  const __bf16* apl = a_lo + (size_t)arow * DDIM;
  v16bf ah[8], al[8];
#pragma unroll
  for (int kk = 0; kk < 8; ++kk) {
    const int k0 = kk * 32 + abase;
    v8bf h0 = *(const v8bf*)(aph + k0);
    v8bf h1 = *(const v8bf*)(aph + k0 + 8);
    ah[kk] = __builtin_shufflevector(h0, h1, 0,1,2,3,4,5,6,7,8,9,10,11,12,13,14,15);
    v8bf l0 = *(const v8bf*)(apl + k0);
    v8bf l1 = *(const v8bf*)(apl + k0 + 8);
    al[kk] = __builtin_shufflevector(l0, l1, 0,1,2,3,4,5,6,7,8,9,10,11,12,13,14,15);
  }

  float lsum[8], psum[8];
#pragma unroll
  for (int k = 0; k < 8; ++k) { lsum[k] = 0.0f; psum[k] = 0.0f; }
  float cnte = 0.0f;

  // this wave's contiguous slice of column tiles
  const int ct0 = wave * (NCT / 4);
  const int ct1 = ct0 + (NCT / 4);
  for (int ct = ct0; ct < ct1; ++ct) {
    const int col   = ct * 16 + l16;
    const int bbase = half * 16;       // contiguous K order (matches A): lane half selects K 0/16
    const __bf16* bph = b_hi + (size_t)col * DDIM + bbase;
    const __bf16* bpl = b_lo + (size_t)col * DDIM + bbase;
    // Unconditional speculative prefetch, 2 tiles ahead, both streams.
    // global_prefetch_b8 never faults (failed translations dropped) and does
    // not touch LOADcnt, so no guard / exec manipulation is needed.
    __builtin_prefetch(b_hi + (size_t)(col + 32) * DDIM, 0, 0);
    __builtin_prefetch(b_lo + (size_t)(col + 32) * DDIM, 0, 0);

    v8f acc;
#pragma unroll
    for (int k = 0; k < 8; ++k) acc[k] = 0.0f;

#pragma unroll
    for (int kk = 0; kk < 8; ++kk) {
      v16bf bh = *(const v16bf*)(bph + kk * 32);
      v16bf bl = *(const v16bf*)(bpl + kk * 32);
      acc = __builtin_amdgcn_wmma_f32_16x16x32_bf16(false, ah[kk], false, bh,
                                                    (short)0, acc, false, false);
      acc = __builtin_amdgcn_wmma_f32_16x16x32_bf16(false, ah[kk], false, bl,
                                                    (short)0, acc, false, false);
      acc = __builtin_amdgcn_wmma_f32_16x16x32_bf16(false, al[kk], false, bh,
                                                    (short)0, acc, false, false);
    }

    const float cv  = c_valid[col];
    const bool  pos = (ct >> 5) == myclass;   // class span 512 = 32 col-tiles
#pragma unroll
    for (int k = 0; k < 8; ++k) {
      const float s = acc[k] * INVT;          // unit dots -> s <= 10 always
      lsum[k] += __expf(s - 10.0f) * cv;      // fixed-shift softmax; cv masks
      if (pos) psum[k] += s * cv;
    }
    if (pos) cnte += cv;
  }

  // Row r of C lives in one 16-lane group (rows 0-7: lanes 0-15, 8-15: 16-31):
  // reduce partials across that group.
#pragma unroll
  for (int k = 0; k < 8; ++k) {
    float L = lsum[k], P = psum[k];
    for (int off = 1; off < 16; off <<= 1) {
      L += __shfl_xor(L, off, 32);
      P += __shfl_xor(P, off, 32);
    }
    lsum[k] = L; psum[k] = P;
  }
  float CN = cnte;
  for (int off = 1; off < 16; off <<= 1) CN += __shfl_xor(CN, off, 32);

  // stash per-wave partials; fixed shift means merging is a plain add
  if (l16 == 0) {
#pragma unroll
    for (int k = 0; k < 8; ++k) {
      s_l[wave][k + 8 * half] = lsum[k];
      s_p[wave][k + 8 * half] = psum[k];
    }
    if (lane == 0) s_c[wave] = CN;
  }
  __syncthreads();

  if (threadIdx.x < 16) {
    const int rr  = threadIdx.x;
    const int row = rt + rr;
    float L = 0.0f, P = 0.0f, C = 0.0f;
#pragma unroll
    for (int w = 0; w < 4; ++w) {
      L += s_l[w][rr];
      P += s_p[w][rr];
      C += s_c[w];
    }
    const float lse = 10.0f + __logf(L);
    const bool  va  = (C > 0.0f) && (a_valid[row] > 0.5f);
    mlpp[row] = va ? (P - C * lse) / fmaxf(C, 1.0f) : 0.0f;
    varr[row] = va ? 1.0f : 0.0f;
  }
}

// ---------------------------------------------------------------------------
// K4: deterministic fixed-order reduction -> scalar loss.
// ---------------------------------------------------------------------------
__global__ __launch_bounds__(256) void mbcl_finalize(const float* __restrict__ mlpp,
                                                     const float* __restrict__ varr,
                                                     float* __restrict__ out) {
  __shared__ float r1[256];
  __shared__ float r2[256];
  const int t = threadIdx.x;
  float s = 0.0f, c = 0.0f;
  for (int i = t; i < AROWS; i += 256) { s += mlpp[i]; c += varr[i]; }
  r1[t] = s; r2[t] = c; __syncthreads();
  for (int k = 128; k > 0; k >>= 1) {
    if (t < k) { r1[t] += r1[t + k]; r2[t] += r2[t + k]; }
    __syncthreads();
  }
  if (t == 0) out[0] = -r1[0] / fmaxf(r2[0], 1.0f);
}

// ---------------------------------------------------------------------------
extern "C" void kernel_launch(void* const* d_in, const int* in_sizes, int n_in,
                              void* d_out, int out_size, void* d_ws, size_t ws_size,
                              hipStream_t stream) {
  const float* mainp = (const float*)d_in[0];
  const int*   mgt   = (const int*)d_in[1];
  const float* auxp  = (const float*)d_in[2];
  const int*   agt   = (const int*)d_in[3];
  const float* bank  = (const float*)d_in[4];
  float* out = (float*)d_out;

  // workspace carve (all chunks 256B-aligned)
  char* w = (char*)d_ws;
  int*    order   = (int*)w;     w += (size_t)AROWS * 4;          // 40960
  int*    counts  = (int*)w;     w += 256;
  float*  a_valid = (float*)w;   w += (size_t)AROWS * 4;
  float*  c_valid = (float*)w;   w += (size_t)AROWS * 4;
  float*  mlpp    = (float*)w;   w += (size_t)AROWS * 4;
  float*  varr    = (float*)w;   w += (size_t)AROWS * 4;
  __bf16* a_hi    = (__bf16*)w;  w += (size_t)AROWS * DDIM * 2;   // 5.25 MB each
  __bf16* a_lo    = (__bf16*)w;  w += (size_t)AROWS * DDIM * 2;
  __bf16* b_hi    = (__bf16*)w;  w += (size_t)AROWS * DDIM * 2;
  __bf16* b_lo    = (__bf16*)w;  w += (size_t)AROWS * DDIM * 2;

  mbcl_select<<<NCLS, 1024, 0, stream>>>(mgt, agt, order, counts);
  mbcl_gather<<<AROWS, 256, 0, stream>>>(mainp, auxp, bank, order, counts,
                                         a_valid, c_valid, a_hi, a_lo, b_hi, b_lo);
  mbcl_gemm_loss<<<AROWS / 16, 128, 0, stream>>>(a_hi, a_lo, b_hi, b_lo,
                                                 a_valid, c_valid, mlpp, varr);
  mbcl_finalize<<<1, 256, 0, stream>>>(mlpp, varr, out);
}
